// ScaledDotProductAttention_3582002725554
// MI455X (gfx1250) — matrix operand order repaired and checked by reading, not compiled, not run on hardware
//
#include <hip/hip_runtime.h>
#include <hip/hip_bf16.h>

typedef __attribute__((ext_vector_type(16))) _Float16 v16h;
typedef __attribute__((ext_vector_type(8)))  _Float16 h8;
typedef __attribute__((ext_vector_type(8)))  float    v8f;

#define ATTN_B 4
#define ATTN_H 16
#define ATTN_S 2048
#define ATTN_D 64
#define PROW 2056            // f16 strip row stride in halves (2048+8): 4112B rows -> 16B aligned, 4-bank lane skew
#define MASKED_F16 (-60000.0f)   // representable in f16; exp() underflows to exactly 0

// One block = one (b,h) x 16-row query tile. 256 threads = 8 wave32. ~70KB LDS -> 4 WGs/WGP.
__global__ __launch_bounds__(256)
void sdpa_fused_kernel(const float* __restrict__ Q,
                       const float* __restrict__ K,
                       const float* __restrict__ V,
                       const int*   __restrict__ mask,
                       const int*   __restrict__ scale_p,
                       float* __restrict__ out,     // [B,H,S,D]
                       float* __restrict__ attn) {  // [B,H,S,S]
    constexpr int S = ATTN_S, D = ATTN_D, H = ATTN_H;

    const int qt   = blockIdx.x;
    const int hh   = blockIdx.y;
    const int bb   = blockIdx.z;
    const int bh   = bb * H + hh;
    const int q0   = qt << 4;

    const int tid  = threadIdx.x;
    const int lane = tid & 31;
    const int wave = tid >> 5;
    const int lo   = lane & 15;    // M/N index within 16
    const int hi   = lane >> 4;    // half-wave selector

    const float* Qb = Q + (size_t)bh * S * D + (size_t)q0 * D;
    const float* Kb = K + (size_t)bh * S * D;
    const float* Vb = V + (size_t)bh * S * D;
    const int*   Mb = mask + (size_t)bh * S * S + (size_t)q0 * S;
    float*       Ob = out  + (size_t)bh * S * D + (size_t)q0 * D;
    float*       Pb = attn + (size_t)bh * S * S + (size_t)q0 * S;

    extern __shared__ char smem_raw[];
    _Float16* sP   = (_Float16*)smem_raw;                    // [16][PROW] f16 score/prob strip
    float*    sRed = (float*)(smem_raw + 16 * PROW * 2);     // [4][16][16] cross-wave reduce

    const float inv_scale = 1.0f / (float)(*scale_p);

    // ---- Stage 1: Q A-fragments (16x64 -> two 16x32 f16 fragments), per-wave ----
    v16h aQ0, aQ1;
    {
        const float* qrow = Qb + lo * D;
        #pragma unroll
        for (int e = 0; e < 16; ++e) {
            const int d = (hi << 4) + e;
            aQ0[e] = (_Float16)qrow[d];
            aQ1[e] = (_Float16)qrow[d + 32];
        }
    }

    // ---- Stage 2: scores = QK^T/scale, mask, store f16 to LDS. 128 key-tiles, 16/wave ----
    for (int it = 0; it < 16; ++it) {
        const int key0 = (wave + (it << 3)) << 4;

        // B-fragment: B[d][n] = K[key0+n][d]; 16 contiguous floats per lane per chunk (b128 loads)
        const float4* kv = (const float4*)(Kb + (size_t)(key0 + lo) * D + (hi << 4));
        float f0[16], f1[16];
        *(float4*)&f0[0]  = kv[0];  *(float4*)&f0[4]  = kv[1];
        *(float4*)&f0[8]  = kv[2];  *(float4*)&f0[12] = kv[3];
        *(float4*)&f1[0]  = kv[8];  *(float4*)&f1[4]  = kv[9];
        *(float4*)&f1[8]  = kv[10]; *(float4*)&f1[12] = kv[11];
        v16h bK0, bK1;
        #pragma unroll
        for (int e = 0; e < 16; ++e) { bK0[e] = (_Float16)f0[e]; bK1[e] = (_Float16)f1[e]; }

        v8f c = {};
        c = __builtin_amdgcn_wmma_f32_16x16x32_f16(false, aQ0, false, bK0, (short)0, c, false, false);
        c = __builtin_amdgcn_wmma_f32_16x16x32_f16(false, aQ1, false, bK1, (short)0, c, false, false);

        #pragma unroll
        for (int r = 0; r < 8; ++r) {
            const int m = r + (hi << 3);      // q row in tile
            float s = c[r] * inv_scale;
            if (Mb[(size_t)m * S + key0 + lo] == 0) s = MASKED_F16;
            sP[m * PROW + key0 + lo] = (_Float16)s;
        }
    }
    __syncthreads();

    // ---- Stage 3: row softmax in f16 strip (2 rows/wave), attention out as f32 float4s ----
    #pragma unroll
    for (int rr = 0; rr < 2; ++rr) {
        const int m = wave * 2 + rr;
        _Float16* row = sP + m * PROW;

        float mx = -3.4e38f;
        #pragma unroll
        for (int i = 0; i < 8; ++i) {
            const h8 v = *(const h8*)(row + lane * 8 + i * 256);
            #pragma unroll
            for (int e = 0; e < 8; ++e) mx = fmaxf(mx, (float)v[e]);
        }
        #pragma unroll
        for (int off = 16; off > 0; off >>= 1) mx = fmaxf(mx, __shfl_xor(mx, off, 32));

        float sum = 0.0f;
        #pragma unroll
        for (int i = 0; i < 8; ++i) {
            _Float16* p = row + lane * 8 + i * 256;
            h8 v = *(const h8*)p;
            #pragma unroll
            for (int e = 0; e < 8; ++e) {
                const float ex = __expf((float)v[e] - mx);
                sum += ex;
                v[e] = (_Float16)ex;
            }
            *(h8*)p = v;                         // keep un-normalized exp in LDS
        }
        #pragma unroll
        for (int off = 16; off > 0; off >>= 1) sum += __shfl_xor(sum, off, 32);
        const float invs = 1.0f / sum;

        float* prow = Pb + (size_t)m * S;
        #pragma unroll
        for (int i = 0; i < 8; ++i) {
            const int c0 = lane * 8 + i * 256;
            _Float16* p = row + c0;
            h8 v = *(const h8*)p;
            float4 o0, o1;
            o0.x = (float)v[0] * invs; o0.y = (float)v[1] * invs;
            o0.z = (float)v[2] * invs; o0.w = (float)v[3] * invs;
            o1.x = (float)v[4] * invs; o1.y = (float)v[5] * invs;
            o1.z = (float)v[6] * invs; o1.w = (float)v[7] * invs;
            #pragma unroll
            for (int e = 0; e < 8; ++e) v[e] = (_Float16)((float)v[e] * invs);
            *(h8*)p = v;                          // normalized P (f16) for stage 4
            *(float4*)(prow + c0)     = o0;       // coalesced 1KB/wave/step attention write
            *(float4*)(prow + c0 + 4) = o1;
        }
    }
    __syncthreads();

    // ---- Stage 4: out = P(16x2048) x V(2048x64); wave -> (ntile, k-half) ----
    const int nt = wave & 3;
    const int kh = wave >> 2;
    const _Float16* arow = sP + lo * PROW;
    v8f acc = {};
    for (int i = 0; i < 32; ++i) {
        const int kbase = kh * 1024 + (i << 5);

        // A-fragment: two contiguous 16B LDS reads per lane (ds_load_b128), bank-conflict-free
        const h8 p0 = *(const h8*)(arow + kbase + (hi << 3));
        const h8 p1 = *(const h8*)(arow + kbase + 16 + (hi << 3));
        v16h aP;
        #pragma unroll
        for (int e = 0; e < 8; ++e) { aP[e] = p0[e]; aP[e + 8] = p1[e]; }

        v16h bV;
        #pragma unroll
        for (int e = 0; e < 16; ++e) {
            const int kr = kbase + ((e >> 3) << 4) + (hi << 3) + (e & 7);
            bV[e] = (_Float16)Vb[(size_t)kr * D + nt * 16 + lo];
        }
        acc = __builtin_amdgcn_wmma_f32_16x16x32_f16(false, aP, false, bV, (short)0, acc, false, false);
    }

    if (kh == 1) {
        #pragma unroll
        for (int r = 0; r < 8; ++r)
            sRed[(nt * 16 + r + (hi << 3)) * 16 + lo] = acc[r];
    }
    __syncthreads();
    if (kh == 0) {
        #pragma unroll
        for (int r = 0; r < 8; ++r) {
            const int m = r + (hi << 3);
            Ob[(size_t)m * D + nt * 16 + lo] = acc[r] + sRed[(nt * 16 + m) * 16 + lo];
        }
    }
}

extern "C" void kernel_launch(void* const* d_in, const int* in_sizes, int n_in,
                              void* d_out, int out_size, void* d_ws, size_t ws_size,
                              hipStream_t stream) {
    (void)in_sizes; (void)n_in; (void)d_ws; (void)ws_size; (void)out_size;
    const float* Q     = (const float*)d_in[0];
    const float* K     = (const float*)d_in[1];
    const float* V     = (const float*)d_in[2];
    const int*   mask  = (const int*)d_in[3];
    const int*   scale = (const int*)d_in[4];

    float* out  = (float*)d_out;                                      // [B,H,S,D]
    float* attn = out + (size_t)ATTN_B * ATTN_H * ATTN_S * ATTN_D;    // [B,H,S,S]

    const size_t lds = (size_t)16 * PROW * sizeof(_Float16)    // f16 score/prob strip (~64KB)
                     + (size_t)4 * 16 * 16 * sizeof(float);    // reduce buffer (4KB)

    dim3 grid(ATTN_S / 16, ATTN_H, ATTN_B);
    dim3 block(256);
    sdpa_fused_kernel<<<grid, block, lds, stream>>>(Q, K, V, mask, scale, out, attn);
}
